// EdgeConv_61194694033723
// MI455X (gfx1250) — matrix operand order repaired and checked by reading, not compiled, not run on hardware
//
#include <hip/hip_runtime.h>

#define BATCH   16
#define CH      32
#define NPTS    2048
#define HID     64
#define KNB     16
#define EPSV    1e-5f

typedef __attribute__((ext_vector_type(16))) _Float16 v16h;
typedef __attribute__((ext_vector_type(8)))  _Float16 v8h;
typedef __attribute__((ext_vector_type(8)))  float    v8f;

// LDS pitches (in halves / floats) chosen for bank-conflict-free strides
#define PITCH_H 72      // 144B = 36 dwords, gcd(36,64)=4 -> 16 distinct banks over 16 lanes
#define PITCH_F 36      // knn column tile pitch in floats

// ---------------------------------------------------------------------------
// Kernel 1: fold BN into alpha/beta, convert weights to f16
// coefs layout: [0:64) alpha0 [64:128) beta0 [128:192) alpha1 [192:256) beta1
// ---------------------------------------------------------------------------
__global__ __launch_bounds__(256) void prep_kernel(
    const float* __restrict__ W0, const float* __restrict__ b0,
    const float* __restrict__ g0, const float* __restrict__ be0,
    const float* __restrict__ m0, const float* __restrict__ v0,
    const float* __restrict__ W1, const float* __restrict__ b1,
    const float* __restrict__ g1, const float* __restrict__ be1,
    const float* __restrict__ m1, const float* __restrict__ v1,
    float* __restrict__ coefs, _Float16* __restrict__ W0h, _Float16* __restrict__ W1h)
{
    int t = threadIdx.x;
    if (t < 64) {
        float a = g0[t] * rsqrtf(v0[t] + EPSV);
        coefs[t]      = a;
        coefs[64 + t] = (b0[t] - m0[t]) * a + be0[t];
    } else if (t < 128) {
        int u = t - 64;
        float a = g1[u] * rsqrtf(v1[u] + EPSV);
        coefs[128 + u] = a;
        coefs[192 + u] = (b1[u] - m1[u]) * a + be1[u];
    }
    for (int i = t; i < HID * 2 * CH; i += 256) {
        W0h[i] = (_Float16)W0[i];
        W1h[i] = (_Float16)W1[i];
    }
}

// ---------------------------------------------------------------------------
// Kernel 2: squared norms per point
// ---------------------------------------------------------------------------
__global__ __launch_bounds__(256) void sq_kernel(const float* __restrict__ x,
                                                 float* __restrict__ sq)
{
    int i = blockIdx.x * 256 + threadIdx.x;   // i = b*NPTS + n
    int b = i >> 11, n = i & (NPTS - 1);
    const float* xb = x + b * CH * NPTS;
    float s = 0.f;
#pragma unroll
    for (int c = 0; c < CH; ++c) { float v = xb[c * NPTS + n]; s = fmaf(v, v, s); }
    sq[i] = s;
}

// ---------------------------------------------------------------------------
// Kernel 3: top-16 neighbors per point (register insertion sort)
// grid: (NPTS/128, BATCH), 128 threads (one row each)
// ---------------------------------------------------------------------------
__global__ __launch_bounds__(128) void knn_kernel(const float* __restrict__ x,
                                                  const float* __restrict__ sq,
                                                  int* __restrict__ idx)
{
    __shared__ float colX[128 * PITCH_F];
    __shared__ float sqc[128];

    const int b = blockIdx.y;
    const int n = blockIdx.x * 128 + threadIdx.x;
    const float* xb = x + b * CH * NPTS;

    float rowx[CH];
#pragma unroll
    for (int c = 0; c < CH; ++c) rowx[c] = xb[c * NPTS + n];
    const float sqr = sq[b * NPTS + n];

    float bd[KNB]; int bi[KNB];
#pragma unroll
    for (int j = 0; j < KNB; ++j) { bd[j] = -3.0e38f; bi[j] = 0; }

    for (int m0c = 0; m0c < NPTS; m0c += 128) {
        __syncthreads();
        const int t = threadIdx.x;
#pragma unroll
        for (int c = 0; c < CH; ++c) colX[t * PITCH_F + c] = xb[c * NPTS + m0c + t];
        sqc[t] = sq[b * NPTS + m0c + t];
        __syncthreads();

        for (int mm = 0; mm < 128; ++mm) {
            float dot = 0.f;
#pragma unroll
            for (int c = 0; c < CH; ++c) dot = fmaf(rowx[c], colX[mm * PITCH_F + c], dot);
            float nd = 2.f * dot - sqr - sqc[mm];
            if (nd > bd[KNB - 1]) {
                float cd = nd; int ci = m0c + mm;
#pragma unroll
                for (int j = 0; j < KNB; ++j) {          // bubble-insert, ties keep earlier
                    bool sw = cd > bd[j];
                    float td = sw ? bd[j] : cd;  int ti = sw ? bi[j] : ci;
                    bd[j] = sw ? cd : bd[j];     bi[j] = sw ? ci : bi[j];
                    cd = td; ci = ti;
                }
            }
        }
    }
    int* op = idx + (b * NPTS + n) * KNB;
#pragma unroll
    for (int j = 0; j < KNB; ++j) op[j] = bi[j];
}

// ---------------------------------------------------------------------------
// GEMM layer helper: D(64 x 256) = W(64x64) * B(64x256), BN + ReLU epilogue.
// Wave wv owns 8 of the 64 (mt,ct) 16x16 tiles; K=64 -> 2 WMMA f16 steps.
// Operand layouts per CDNA5 ISA 7.12.2 (wave32).
// ---------------------------------------------------------------------------
template <bool TO_LDS>
__device__ __forceinline__ void gemm64(
    const _Float16* __restrict__ sW, const _Float16* __restrict__ sB,
    const float* __restrict__ sAl, const float* __restrict__ sBe,
    _Float16* __restrict__ sO, float* __restrict__ gO, int gColBase,
    int wv, int lhalf, int hi)
{
#pragma unroll
    for (int i = 0; i < 8; ++i) {
        const int tid = wv * 8 + i;
        const int mt  = tid >> 4;        // M tile (0..3)
        const int ct  = tid & 15;        // col tile (0..15)
        const int col = ct * 16 + lhalf; // D col for this lane
        const int M   = mt * 16 + lhalf; // A row for this lane
        v8f acc = {};
#pragma unroll
        for (int ks = 0; ks < 2; ++ks) {
            // A 16x32 f16: lane<16 -> K[0..7],K[16..23]; lane>=16 -> +8
            v8h a0 = *(const v8h*)(sW + M * PITCH_H + ks * 32 + hi * 8);
            v8h a1 = *(const v8h*)(sW + M * PITCH_H + ks * 32 + hi * 8 + 16);
            // B 32x16 f16: lane<16 -> K[0..7],K[16..23]; lane>=16 -> +8 (contig per col)
            const v8h* bp = (const v8h*)(sB + col * PITCH_H + ks * 32 + hi * 8);
            v8h b0v = bp[0], b1v = bp[2];
            v16h av, bv;
#pragma unroll
            for (int q = 0; q < 8; ++q) {
                av[q] = a0[q]; av[q + 8] = a1[q];
                bv[q] = b0v[q]; bv[q + 8] = b1v[q];
            }
            acc = __builtin_amdgcn_wmma_f32_16x16x32_f16(
                false, av, false, bv, (short)0, acc, false, false);
        }
        const int chb = mt * 16 + hi * 8;   // D row base for this lane
        if (TO_LDS) {
            v8h hv;
#pragma unroll
            for (int r = 0; r < 8; ++r) {
                float y = acc[r] * sAl[chb + r] + sBe[chb + r];
                y = y > 0.f ? y : 0.f;
                hv[r] = (_Float16)y;
            }
            *(v8h*)(sO + col * PITCH_H + chb) = hv;   // 16B aligned, pitch-72 cols
        } else {
#pragma unroll
            for (int r = 0; r < 8; ++r) {
                float y = acc[r] * sAl[chb + r] + sBe[chb + r];
                y = y > 0.f ? y : 0.f;
                gO[(chb + r) * (NPTS * KNB) + gColBase + col] = y;
            }
        }
    }
}

// ---------------------------------------------------------------------------
// Kernel 4: fused feature-build + conv1(BN,ReLU) + conv2(BN,ReLU)
// 1 WG = 1 batch x 16 points x 16 neighbors = 256 columns, 256 threads (8 waves)
// ---------------------------------------------------------------------------
__global__ __launch_bounds__(256) void fused_kernel(
    const float* __restrict__ x, const int* __restrict__ idx,
    const _Float16* __restrict__ W0h, const _Float16* __restrict__ W1h,
    const float* __restrict__ coefs, float* __restrict__ out)
{
    extern __shared__ char smem[];
    _Float16* sW0 = (_Float16*)smem;            // 64 x PITCH_H
    _Float16* sW1 = sW0 + HID * PITCH_H;
    _Float16* sF  = sW1 + HID * PITCH_H;        // 256 cols x PITCH_H (col-major feat)
    _Float16* sH  = sF  + 256 * PITCH_H;        // 256 cols x PITCH_H (hidden)
    float*    sC  = (float*)(sH + 256 * PITCH_H); // 256 floats: a0,b0,a1,b1

    const int t  = threadIdx.x;
    const int b  = blockIdx.x >> 7;             // 128 point-tiles per batch
    const int nt = blockIdx.x & 127;
    const int n0 = nt * 16;
    const float* xb = x + b * CH * NPTS;

    // stage weights (pitch-adjust 64 -> 72) and BN coefs
    {
        const int row = t >> 2, seg = (t & 3) * 16;
        const uint4* s0 = (const uint4*)(W0h + row * 64 + seg);
        const uint4* s1 = (const uint4*)(W1h + row * 64 + seg);
        uint4* d0 = (uint4*)(sW0 + row * PITCH_H + seg);
        uint4* d1 = (uint4*)(sW1 + row * PITCH_H + seg);
        d0[0] = s0[0]; d0[1] = s0[1];
        d1[0] = s1[0]; d1[1] = s1[1];
        sC[t] = coefs[t];
    }

    // build edge features for this thread's column: ch[0:32)=nbr-ctr, [32:64)=ctr
    {
        const int ln = t >> 4, k = t & 15;
        const int n  = n0 + ln;
        const int m  = idx[(b * NPTS + n) * KNB + k];
        _Float16* fc = sF + t * PITCH_H;
#pragma unroll
        for (int c = 0; c < CH; ++c) {
            float ce = xb[c * NPTS + n];
            float nb = xb[c * NPTS + m];
            fc[c]      = (_Float16)(nb - ce);
            fc[CH + c] = (_Float16)ce;
        }
    }
    __syncthreads();

    const int lane = t & 31, wv = t >> 5;
    const int lhalf = lane & 15, hi = lane >> 4;

    // layer 1: h = relu(a0 * (W0 f) + b0) -> LDS f16
    gemm64<true>(sW0, sF, sC, sC + 64, sH, nullptr, 0, wv, lhalf, hi);
    __syncthreads();

    // layer 2: out = relu(a1 * (W1 h) + b1) -> global f32
    float* gO = out + b * HID * NPTS * KNB;
    gemm64<false>(sW1, sH, sC + 128, sC + 192, nullptr, gO, n0 * KNB, wv, lhalf, hi);
}

// ---------------------------------------------------------------------------
extern "C" void kernel_launch(void* const* d_in, const int* in_sizes, int n_in,
                              void* d_out, int out_size, void* d_ws, size_t ws_size,
                              hipStream_t stream)
{
    const float* x   = (const float*)d_in[0];
    const float* W0  = (const float*)d_in[1];
    const float* b0  = (const float*)d_in[2];
    const float* g0  = (const float*)d_in[3];
    const float* be0 = (const float*)d_in[4];
    const float* m0  = (const float*)d_in[5];
    const float* v0  = (const float*)d_in[6];
    const float* W1  = (const float*)d_in[7];
    const float* b1  = (const float*)d_in[8];
    const float* g1  = (const float*)d_in[9];
    const float* be1 = (const float*)d_in[10];
    const float* m1  = (const float*)d_in[11];
    const float* v1  = (const float*)d_in[12];
    float* out = (float*)d_out;

    char* ws = (char*)d_ws;
    float*    sq    = (float*)ws;                       // 131072 B
    int*      idx   = (int*)(ws + 131072);              // 2097152 B
    float*    coefs = (float*)(ws + 2228224);           // 1024 B
    _Float16* W0h   = (_Float16*)(ws + 2229248);        // 8192 B
    _Float16* W1h   = W0h + HID * 2 * CH;               // 8192 B

    prep_kernel<<<1, 256, 0, stream>>>(W0, b0, g0, be0, m0, v0,
                                       W1, b1, g1, be1, m1, v1,
                                       coefs, W0h, W1h);
    sq_kernel<<<(BATCH * NPTS) / 256, 256, 0, stream>>>(x, sq);
    knn_kernel<<<dim3(NPTS / 128, BATCH), 128, 0, stream>>>(x, sq, idx);

    const size_t smem = (size_t)(2 * HID * PITCH_H + 2 * 256 * PITCH_H) * sizeof(_Float16)
                        + 256 * sizeof(float);          // 93,184 B
    fused_kernel<<<BATCH * (NPTS / 16), 256, smem, stream>>>(x, idx, W0h, W1h, coefs, out);
}
